// MultiEncoderBlock_10651518894283
// MI455X (gfx1250) — matrix operand order repaired and checked by reading, not compiled, not run on hardware
//
#include <hip/hip_runtime.h>
#include <hip/hip_bf16.h>

// ---------------- problem constants ----------------
constexpr int L     = 4096;   // 16*16*16
constexpr int BSZ   = 2;
constexpr int DM    = 128;    // d_model
constexpr int DI    = 256;    // d_inner
constexpr int DS    = 16;     // d_state
constexpr int DTR   = 8;      // dt_rank
constexpr int NDIR  = 6;
constexpr int XZ    = 2 * DI; // 512
constexpr int NXP   = 40;     // dt_rank + 2*d_state
constexpr int NXPP  = 48;     // padded to multiple of 16
constexpr int CHUNK = 128;
constexpr int NCHUNK = L / CHUNK; // 32
constexpr int LDS_PAD = 8;    // bf16 elements (16 B) of per-row LDS padding

typedef __attribute__((ext_vector_type(16))) __bf16 v16bf;
typedef __attribute__((ext_vector_type(8)))  float  v8f;
typedef unsigned int v4u __attribute__((ext_vector_type(4)));
typedef int          v4i __attribute__((ext_vector_type(4)));
typedef int          v8i __attribute__((ext_vector_type(8)));

// TDM availability (device pass only; host pass uses fallback body).
#if defined(__HIP_DEVICE_COMPILE__) && defined(__has_builtin)
#  if __has_builtin(__builtin_amdgcn_tensor_load_to_lds) && \
      __has_builtin(__builtin_amdgcn_s_wait_tensorcnt)
#    define USE_TDM 1
#  endif
#endif
#ifndef USE_TDM
#  define USE_TDM 0
#endif

// ---------------- direction index maps ----------------
// seq position l (direction dir) -> spatial flat index s = (d*16+h)*16+w
__device__ __forceinline__ int seq_to_spatial(int dir, int l) {
  if (dir >= 3) l = L - 1 - l;
  int a  = l >> 8;
  int b2 = (l >> 4) & 15;
  int c3 = l & 15;
  int d, h, w;
  switch (dir % 3) {
    case 0:  d = a;  h = b2; w = c3;     break;  // natural
    case 1:  d = a;  h = c3; w = b2;     break;  // swap h<->w
    default: d = c3; h = b2; w = 15 - a; break;  // rot90 in (d,w)
  }
  return (d * 16 + h) * 16 + w;
}

// ---------------- weight conversion (per dir) ----------------
__global__ void convert_weights(const float* __restrict__ W_in,
                                const float* __restrict__ W_xproj,
                                const float* __restrict__ W_out,
                                __bf16* __restrict__ win,
                                __bf16* __restrict__ wxp,
                                __bf16* __restrict__ wout, int dir) {
  int i = blockIdx.x * blockDim.x + threadIdx.x;
  if (i < XZ * DM) win[i] = (__bf16)W_in[(size_t)dir * XZ * DM + i];
  if (i < NXPP * DI) {
    int r = i / DI, c = i % DI;
    wxp[i] = (r < NXP) ? (__bf16)W_xproj[((size_t)dir * NXP + r) * DI + c]
                       : (__bf16)0.f;
  }
  if (i < DM * DI) wout[i] = (__bf16)W_out[(size_t)dir * DM * DI + i];
}

// ---------------- pack input into sequence order (bf16) ----------------
__global__ void pack_x(const float* __restrict__ in, __bf16* __restrict__ xb,
                       int dir) {
  int idx = blockIdx.x * blockDim.x + threadIdx.x;   // (b, l, c)
  if (idx >= BSZ * L * DM) return;
  int c = idx & (DM - 1);
  int l = (idx >> 7) & (L - 1);
  int b = idx >> 19;
  int s = seq_to_spatial(dir, l);
  xb[idx] = (__bf16)in[((size_t)b * DM + c) * L + s];
}

// -------- bf16 WMMA GEMM with TDM-staged LDS weight panel ----------------
// C[M,N] = A[M,K] * W[N,K]^T.
// Block owns the N-panel [blockIdx.y*16*TN, +16*TN) staged to LDS by the
// Tensor Data Mover (hardware row padding of 16B to dodge bank conflicts).
// Each wave register-blocks MT M-tiles x TN N-tiles (MT*TN wmma per K-step).
template <int TN, int MT>
__global__ void gemm_bf16_wmma(const __bf16* __restrict__ A,
                               const __bf16* __restrict__ W,
                               float* __restrict__ Cout,
                               int M, int N, int K) {
  extern __shared__ __bf16 sW[];       // (16*TN) rows x (K + LDS_PAD)
  const int lane = threadIdx.x & 31;
  const int wave = threadIdx.x >> 5;
  const int n0   = blockIdx.y * (16 * TN);
  const int KP   = K + LDS_PAD;
  const int rows = 16 * TN;

#if USE_TDM
  if (threadIdx.x < 32) {              // wave 0 issues the TDM descriptor
    unsigned lds = (unsigned)(size_t)(&sW[0]);
    unsigned long long ga =
        (unsigned long long)(size_t)(W + (size_t)n0 * K);
    v4u g0;
    g0[0] = 1u;                                          // count=1, user mode
    g0[1] = lds;                                         // lds_addr (bytes)
    g0[2] = (unsigned)(ga & 0xFFFFFFFFu);                // global_addr[31:0]
    g0[3] = (unsigned)((ga >> 32) & 0x01FFFFFFu)         // global_addr[56:32]
            | (2u << 30);                                // type = 2 (image)
    // pad_interval: pad after K elements = K/2 DWORDs -> code log2(K)-2
    const unsigned pad_iv = (unsigned)(__builtin_ctz((unsigned)K) - 2);
    v8i g1;
    g1[0] = (int)((1u << 16)            // data_size = 1 (2-byte elements)
                  | (1u << 20)          // pad_enable
                  | (pad_iv << 22)      // pad_interval
                  | (3u << 25));        // pad_amount code 3 = 4 DWORDs (16B)
    g1[1] = (int)((unsigned)(K & 0xFFFF) << 16);     // tensor_dim0[15:0]
    g1[2] = (int)(((unsigned)rows & 0xFFFF) << 16);  // tensor_dim1[15:0]
    g1[3] = (int)((unsigned)K << 16);                // tile_dim0
    g1[4] = rows;                                    // tile_dim1
    g1[5] = K;                                       // tensor_dim0_stride lo
    g1[6] = 0;
    g1[7] = 0;
    v4i gz = {0, 0, 0, 0};
#if __clang_major__ >= 23
    v8i gz8 = {0, 0, 0, 0, 0, 0, 0, 0};
    __builtin_amdgcn_tensor_load_to_lds(g0, g1, gz, gz, gz8, 0);
#else
    __builtin_amdgcn_tensor_load_to_lds(g0, g1, gz, gz, 0);
#endif
    __builtin_amdgcn_s_wait_tensorcnt(0);
  }
#else
  for (int i = threadIdx.x; i < rows * K; i += blockDim.x) {
    int r = i / K, c = i - r * K;
    sW[r * KP + c] = W[(size_t)(n0 + r) * K + c];
  }
#endif
  __syncthreads();

  const int m0 = (blockIdx.x * (blockDim.x >> 5) + wave) * (16 * MT);

  v8f acc[MT][TN];
  #pragma unroll
  for (int mt = 0; mt < MT; ++mt)
    #pragma unroll
    for (int t = 0; t < TN; ++t)
      #pragma unroll
      for (int i = 0; i < 8; ++i) acc[mt][t][i] = 0.f;

  // Operand lane layouts used here (wave32, contiguous k, not the ISA table order):
  //  A 16x32: row M = lane%16; elems 0..15 hold K = (lane/16)*16 .. +15.
  //  B 32x16: col N = lane%16; elems 0..15 hold K = (lane/16)*16 .. +15.
  const int mr    = lane & 15;
  const int khalf = (lane >> 4) * 8;
  const int kb    = (lane >> 4) * 16;
  const int ncol  = lane & 15;

  for (int kk = 0; kk < K; kk += 32) {
    v16bf a[MT];
    #pragma unroll
    for (int mt = 0; mt < MT; ++mt) {
      const __bf16* ap = A + (size_t)(m0 + mt * 16 + mr) * K + kk;
      __builtin_prefetch(ap + 32, 0, 1);  // speculative: OOB silently dropped
      #pragma unroll
      for (int i = 0; i < 8; ++i) a[mt][i]     = ap[kb + i];
      #pragma unroll
      for (int i = 0; i < 8; ++i) a[mt][8 + i] = ap[kb + 8 + i];
    }
    #pragma unroll
    for (int t = 0; t < TN; ++t) {
      const __bf16* wp = sW + (size_t)(t * 16 + ncol) * KP + kk + kb;
      v16bf bm;
      #pragma unroll
      for (int i = 0; i < 16; ++i) bm[i] = wp[i];
      #pragma unroll
      for (int mt = 0; mt < MT; ++mt)
        acc[mt][t] = __builtin_amdgcn_wmma_f32_16x16x32_bf16(
            false, a[mt], false, bm, (short)0, acc[mt][t], false, false);
    }
  }
  // D 16x16 f32: lane%16 = N column; 8 rows per lane, M = r + (lane/16)*8.
  const int rbase = (lane >> 4) * 8;
  #pragma unroll
  for (int mt = 0; mt < MT; ++mt)
    #pragma unroll
    for (int t = 0; t < TN; ++t) {
      float* cp = Cout + (size_t)n0 + t * 16 + ncol;
      #pragma unroll
      for (int r = 0; r < 8; ++r)
        cp[(size_t)(m0 + mt * 16 + rbase + r) * N] = acc[mt][t][r];
    }
}

// ---------------- causal depthwise conv (k=4) + SiLU -> bf16 ----------------
__global__ void conv_silu(const float* __restrict__ xz,
                          const float* __restrict__ conv_w,
                          const float* __restrict__ conv_b,
                          __bf16* __restrict__ xcb, int dir) {
  int idx = blockIdx.x * blockDim.x + threadIdx.x;   // (b, l, ch)
  if (idx >= BSZ * L * DI) return;
  int ch = idx & (DI - 1);
  int l  = (idx >> 8) & (L - 1);
  int b  = idx >> 20;
  float acc = conv_b[dir * DI + ch];
  const float* cw = conv_w + ((size_t)dir * DI + ch) * 4;
  #pragma unroll
  for (int j = 0; j < 4; ++j) {
    int ll = l - 3 + j;
    if (ll >= 0) acc += cw[j] * xz[((size_t)b * L + ll) * XZ + ch];
  }
  float s = acc / (1.f + expf(-acc));
  xcb[idx] = (__bf16)s;
}

// ---------------- dt = softplus(dt_lowrank @ W_dt + b_dt); copy B,C --------
__global__ void dt_bc(const float* __restrict__ xdbl,
                      const float* __restrict__ W_dt,
                      const float* __restrict__ b_dt,
                      float* __restrict__ dt, float* __restrict__ BC, int dir) {
  int idx = blockIdx.x * blockDim.x + threadIdx.x;   // (b, l, ch)
  if (idx >= BSZ * L * DI) return;
  int ch = idx & (DI - 1);
  int l  = (idx >> 8) & (L - 1);
  int b  = idx >> 20;
  const float* row = xdbl + ((size_t)b * L + l) * NXPP;
  float v = b_dt[dir * DI + ch];
  const float* wd = W_dt + ((size_t)dir * DI + ch) * DTR;
  #pragma unroll
  for (int r = 0; r < DTR; ++r) v += wd[r] * row[r];
  dt[idx] = (v > 20.f) ? v : log1pf(expf(v));
  if (ch < 2 * DS) BC[((size_t)b * L + l) * 32 + ch] = row[DTR + ch];
}

// ---------------- scan pass 1: local chunk scan + summaries ----------------
__global__ void scan_chunk_local(const float* __restrict__ dt,
                                 const __bf16* __restrict__ xc,
                                 const float* __restrict__ BC,
                                 const float* __restrict__ A_log,
                                 float* __restrict__ y_loc,
                                 float* __restrict__ cA,
                                 float* __restrict__ cH, int dir) {
  __shared__ float sBC[32];
  const int ch = threadIdx.x;               // 256 channels
  const int ci = blockIdx.x & (NCHUNK - 1);
  const int b  = blockIdx.x >> 5;           // NCHUNK = 32
  float aC[DS], h[DS], p[DS];
  const float* al = A_log + ((size_t)dir * DI + ch) * DS;
  #pragma unroll
  for (int n = 0; n < DS; ++n) { aC[n] = -expf(al[n]); h[n] = 0.f; p[n] = 1.f; }
  const int l0 = ci * CHUNK;
  for (int t = 0; t < CHUNK; ++t) {
    size_t pos = (size_t)b * L + (l0 + t);
    if (ch < 32) sBC[ch] = BC[pos * 32 + ch];
    __syncthreads();
    float dtv = dt[pos * DI + ch];
    float xv  = (float)xc[pos * DI + ch];
    float dx  = dtv * xv;
    float y = 0.f;
    #pragma unroll
    for (int n = 0; n < DS; ++n) {
      float a = expf(dtv * aC[n]);
      h[n] = a * h[n] + dx * sBC[n];
      p[n] *= a;
      y += h[n] * sBC[16 + n];
    }
    y_loc[pos * DI + ch] = y;
    __syncthreads();
  }
  size_t co = (((size_t)b * NCHUNK + ci) * DI + ch) * DS;
  #pragma unroll
  for (int n = 0; n < DS; ++n) { cA[co + n] = p[n]; cH[co + n] = h[n]; }
}

// ---------------- scan pass 2: sequential combine across 32 chunks --------
__global__ void scan_combine(const float* __restrict__ cA,
                             const float* __restrict__ cH,
                             float* __restrict__ cHin) {
  int idx = blockIdx.x * blockDim.x + threadIdx.x;   // (b, ch, n)
  if (idx >= BSZ * DI * DS) return;
  int n  = idx & (DS - 1);
  int ch = (idx >> 4) & (DI - 1);
  int b  = idx >> 12;
  float h = 0.f;
  for (int ci = 0; ci < NCHUNK; ++ci) {
    size_t o = (((size_t)b * NCHUNK + ci) * DI + ch) * DS + n;
    cHin[o] = h;
    h = cA[o] * h + cH[o];
  }
}

// ---------------- scan pass 3: correction + D residual + SiLU gate --------
__global__ void scan_fix_gate(const float* __restrict__ dt,
                              const __bf16* __restrict__ xc,
                              const float* __restrict__ BC,
                              const float* __restrict__ A_log,
                              const float* __restrict__ Dparam,
                              const float* __restrict__ xz,
                              const float* __restrict__ y_loc,
                              const float* __restrict__ cHin,
                              __bf16* __restrict__ yg, int dir) {
  __shared__ float sC[16];
  const int ch = threadIdx.x;
  const int ci = blockIdx.x & (NCHUNK - 1);
  const int b  = blockIdx.x >> 5;
  float aC[DS], hin[DS], p[DS];
  const float* al = A_log + ((size_t)dir * DI + ch) * DS;
  size_t co = (((size_t)b * NCHUNK + ci) * DI + ch) * DS;
  #pragma unroll
  for (int n = 0; n < DS; ++n) {
    aC[n] = -expf(al[n]);
    hin[n] = cHin[co + n];
    p[n] = 1.f;
  }
  const float Dv = Dparam[dir * DI + ch];
  const int l0 = ci * CHUNK;
  for (int t = 0; t < CHUNK; ++t) {
    size_t pos = (size_t)b * L + (l0 + t);
    if (ch < 16) sC[ch] = BC[pos * 32 + 16 + ch];
    __syncthreads();
    float dtv = dt[pos * DI + ch];
    float xv  = (float)xc[pos * DI + ch];
    float yc = 0.f;
    #pragma unroll
    for (int n = 0; n < DS; ++n) {
      float a = expf(dtv * aC[n]);
      p[n] *= a;
      yc += sC[n] * p[n] * hin[n];
    }
    float y = y_loc[pos * DI + ch] + yc + xv * Dv;
    float z = xz[pos * XZ + DI + ch];
    float g = z / (1.f + expf(-z));
    yg[pos * DI + ch] = (__bf16)(y * g);
    __syncthreads();
  }
}

// ---------------- final inverse-mapped 6-direction gather ------------------
__global__ void gather_out(const float* __restrict__ y_out,
                           float* __restrict__ out) {
  int idx = blockIdx.x * blockDim.x + threadIdx.x;   // (b, c, s)
  if (idx >= BSZ * DM * L) return;
  int s = idx & (L - 1);
  int c = (idx >> 12) & (DM - 1);
  int b = idx >> 19;
  int a = s >> 8, h = (s >> 4) & 15, w = s & 15;
  int l0 = s;
  int l1 = (a * 16 + w) * 16 + h;
  int l2 = ((15 - w) * 16 + h) * 16 + a;
  int ls[NDIR] = {l0, l1, l2, L - 1 - l0, L - 1 - l1, L - 1 - l2};
  float acc = 0.f;
  #pragma unroll
  for (int d = 0; d < NDIR; ++d)
    acc += y_out[(((size_t)d * BSZ + b) * L + ls[d]) * DM + c];
  out[((size_t)b * DM + c) * L + s] = acc * (1.f / 6.f);
}

// ---------------- host side ----------------
extern "C" void kernel_launch(void* const* d_in, const int* in_sizes, int n_in,
                              void* d_out, int out_size, void* d_ws,
                              size_t ws_size, hipStream_t stream) {
  const float* input   = (const float*)d_in[0];
  const float* W_in    = (const float*)d_in[1];
  const float* conv_w  = (const float*)d_in[2];
  const float* conv_b  = (const float*)d_in[3];
  const float* W_xproj = (const float*)d_in[4];
  const float* W_dt    = (const float*)d_in[5];
  const float* b_dt    = (const float*)d_in[6];
  const float* A_log   = (const float*)d_in[7];
  const float* Dparam  = (const float*)d_in[8];
  const float* W_out   = (const float*)d_in[9];
  float* out = (float*)d_out;

  char* ws = (char*)d_ws;
  size_t off = 0;
  auto alloc = [&](size_t bytes) -> void* {
    void* p = ws + off;
    off += (bytes + 255) & ~(size_t)255;
    return p;
  };
  // persistent across directions
  float* y_out = (float*)alloc((size_t)NDIR * BSZ * L * DM * 4);   // 25.2 MB
  // per-direction reused buffers (all L2-resident, ~50 MB total)
  __bf16* x_bf   = (__bf16*)alloc((size_t)BSZ * L * DM * 2);
  float*  xz     = (float*) alloc((size_t)BSZ * L * XZ * 4);
  __bf16* xc_bf  = (__bf16*)alloc((size_t)BSZ * L * DI * 2);
  float*  xdbl   = (float*) alloc((size_t)BSZ * L * NXPP * 4);
  float*  dtbuf  = (float*) alloc((size_t)BSZ * L * DI * 4);
  float*  BC     = (float*) alloc((size_t)BSZ * L * 32 * 4);
  float*  y_loc  = (float*) alloc((size_t)BSZ * L * DI * 4);
  float*  cA     = (float*) alloc((size_t)BSZ * NCHUNK * DI * DS * 4);
  float*  cH     = (float*) alloc((size_t)BSZ * NCHUNK * DI * DS * 4);
  float*  cHin   = (float*) alloc((size_t)BSZ * NCHUNK * DI * DS * 4);
  __bf16* yg     = (__bf16*)alloc((size_t)BSZ * L * DI * 2);
  __bf16* win_b  = (__bf16*)alloc((size_t)XZ * DM * 2);
  __bf16* wxp_b  = (__bf16*)alloc((size_t)NXPP * DI * 2);
  __bf16* wout_b = (__bf16*)alloc((size_t)DM * DI * 2);
  (void)ws_size; (void)in_sizes; (void)n_in; (void)out_size;

  const int M = BSZ * L;   // 8192 rows
  for (int dir = 0; dir < NDIR; ++dir) {
    convert_weights<<<(XZ * DM + 255) / 256, 256, 0, stream>>>(
        W_in, W_xproj, W_out, win_b, wxp_b, wout_b, dir);
    pack_x<<<(BSZ * L * DM + 255) / 256, 256, 0, stream>>>(input, x_bf, dir);

    // xz = x @ W_in^T : M=8192, N=512, K=128. 8 waves x (2 M-tiles) / block.
    {
      dim3 grid(M / (16 * 2 * 8), XZ / 64);
      size_t smem = (size_t)64 * (DM + LDS_PAD) * sizeof(__bf16);
      gemm_bf16_wmma<4, 2><<<grid, 256, smem, stream>>>(
          x_bf, win_b, xz, M, XZ, DM);
    }

    conv_silu<<<(BSZ * L * DI + 255) / 256, 256, 0, stream>>>(
        xz, conv_w, conv_b, xc_bf, dir);

    // x_dbl = xc @ W_xproj^T : M=8192, N=48 (padded), K=256
    {
      dim3 grid(M / (16 * 2 * 8), 1);
      size_t smem = (size_t)48 * (DI + LDS_PAD) * sizeof(__bf16);
      gemm_bf16_wmma<3, 2><<<grid, 256, smem, stream>>>(
          xc_bf, wxp_b, xdbl, M, NXPP, DI);
    }

    dt_bc<<<(BSZ * L * DI + 255) / 256, 256, 0, stream>>>(
        xdbl, W_dt, b_dt, dtbuf, BC, dir);

    // chunked parallel selective scan
    scan_chunk_local<<<BSZ * NCHUNK, DI, 0, stream>>>(
        dtbuf, xc_bf, BC, A_log, y_loc, cA, cH, dir);
    scan_combine<<<(BSZ * DI * DS + 255) / 256, 256, 0, stream>>>(cA, cH, cHin);
    scan_fix_gate<<<BSZ * NCHUNK, DI, 0, stream>>>(
        dtbuf, xc_bf, BC, A_log, Dparam, xz, y_loc, cHin, yg, dir);

    // y_out[dir] = y_gated @ W_out^T : M=8192, N=128, K=256
    {
      dim3 grid(M / (16 * 2 * 8), DM / 64);
      size_t smem = (size_t)64 * (DI + LDS_PAD) * sizeof(__bf16);
      gemm_bf16_wmma<4, 2><<<grid, 256, smem, stream>>>(
          yg, wout_b, y_out + (size_t)dir * BSZ * L * DM, M, DM, DI);
    }
  }
  gather_out<<<(BSZ * DM * L + 255) / 256, 256, 0, stream>>>(y_out, out);
}
